// CrystalGraphConvNet_42958262894678
// MI455X (gfx1250) — matrix operand order repaired and checked by reading, not compiled, not run on hardware
//
#include <hip/hip_runtime.h>
#include <hip/hip_bf16.h>
#include <cstdint>

// ---------------------------------------------------------------------------
// CrystalGraphConvNet for MI455X (gfx1250, wave32, WMMA f16 16x16x32)
//
// N=100000 atoms, M=12 nbrs, N0=2000 crystals, ORIG=92, F=64, NBR=41,
// G=16, H=128, O=1, NCONV=3.  Conv GEMM K = 2F+NBR = 169 -> padded 192.
// All f16 operand images are padded so staging uses only b128 copies.
// ---------------------------------------------------------------------------

typedef _Float16 half8  __attribute__((ext_vector_type(8)));
typedef _Float16 half16 __attribute__((ext_vector_type(16)));
typedef float    float8 __attribute__((ext_vector_type(8)));

#define NATOM   100000
#define NM      1200000       // N*M rows of the conv GEMM
#define KPAD    192           // 169 padded to 6 chunks of 32
#define NBRP48  48            // nbr_fea f16 row padded 41 -> 48 halves (96 B)
#define APITCH2 200           // LDS pitch (halves) for 192-wide rows; 100 DW, conflict-free
#define APITCH1 104           // LDS pitch (halves) for 96-wide rows; 52 DW, conflict-free
#define EPS_BN  1e-5f

// ---- workspace layout (bytes, 256-aligned) --------------------------------
static constexpr size_t OFF_STATS1 = 0;            // 3*256 f32 (sum[128],sq[128]/layer)
static constexpr size_t OFF_STATS2 = 3072;         // 3*128 f32 (sum[64],sq[64]/layer)
static constexpr size_t OFF_CSUM   = 4608;         // 2000*64 f32
static constexpr size_t OFF_CCNT   = 516608;       // 2000 f32  (zero region ends 524608)
static constexpr size_t ZERO_F32   = 131152;       // floats in zero region
static constexpr size_t OFF_XF32   = 524800;       // 100000*64 f32
static constexpr size_t OFF_XF16   = 26124800;     // 100000*64 f16
static constexpr size_t OFF_NSUM   = 38924800;     // 100000*64 f32
static constexpr size_t OFF_WF16   = 64524800;     // 3*128*192 f16 (padded conv_W)
static constexpr size_t OFF_EMBW   = 64672256;     // 64*96 f16     (padded emb_W)
static constexpr size_t OFF_AF16   = 64684544;     // 100000*96 f16 (padded atom_fea)
static constexpr size_t OFF_NBRP   = 83884544;     // 1200000*48 f16 (padded nbr_fea)
static constexpr size_t OFF_GATED  = 199084544;    // 1200000*128 f16 (blocked layout)
// total ~506.3 MB

// ---- math helpers ---------------------------------------------------------
__device__ __forceinline__ float softplusf(float x) {
    return fmaxf(x, 0.f) + log1pf(expf(-fabsf(x)));
}
__device__ __forceinline__ float sigmoidf(float x) { return 1.f / (1.f + expf(-x)); }

// ---- WMMA fragment helpers -------------------------------------------------
__device__ __forceinline__ half16 cat8(half8 lo, half8 hi) {
    return __builtin_shufflevector(lo, hi, 0,1,2,3,4,5,6,7,8,9,10,11,12,13,14,15);
}
// A fragment (16x32 f16): lane (r = lane&15, h = lane>>4); elems 0..7 = K kb+8h..,
// elems 8..15 = K kb+16+8h..  (CDNA5 ISA 16-bit A layout)
__device__ __forceinline__ half16 load_a_frag(const _Float16* row, int kb, int h) {
    half8 lo = *(const half8*)(row + kb + 8 * h);
    half8 hi = *(const half8*)(row + kb + 16 + 8 * h);
    return cat8(lo, hi);
}
// B fragment (32x16 f16): fixed output column o = nt*16 + (lane&15); elems 0..7 =
// K kc*32+8h.., elems 8..15 = K kc*32+16+8h.. (h = lane>>4).  Source W[o][k] row-major.
__device__ __forceinline__ half16 load_b_frag(const _Float16* W, int pitch, int nt,
                                              int kc, int lane) {
    int o  = nt * 16 + (lane & 15);
    int ks = kc * 32 + ((lane >> 4) << 3);
    const _Float16* p = W + o * pitch + ks;
    return cat8(*(const half8*)p, *(const half8*)(p + 16));
}
__device__ __forceinline__ float8 wmma16(half16 a, half16 b, float8 c) {
    return __builtin_amdgcn_wmma_f32_16x16x32_f16(false, a, false, b, (short)0, c,
                                                  false, false);
}

// ===========================================================================
// K0a: nbr_fea f32 -> f16, rows padded 41 -> 48 halves (16-B aligned rows)
// ===========================================================================
__global__ void cvt_nbr_pad_kernel(const float* __restrict__ src,
                                   _Float16* __restrict__ dst) {
    unsigned i = blockIdx.x * 256 + threadIdx.x;
    if (i >= (unsigned)NM * NBRP48) return;        // 57,600,000
    unsigned nm = i / NBRP48, j = i - nm * NBRP48;
    dst[i] = (j < 41) ? (_Float16)src[(size_t)nm * 41 + j] : (_Float16)0.f;
}

// ===========================================================================
// K0b: atom_fea f32 -> f16, rows padded 92 -> 96 halves
// ===========================================================================
__global__ void cvt_atom_pad_kernel(const float* __restrict__ src,
                                    _Float16* __restrict__ dst) {
    unsigned i = blockIdx.x * 256 + threadIdx.x;
    if (i >= (unsigned)NATOM * 96) return;         // 9,600,000
    unsigned r = i / 96, k = i - r * 96;
    dst[i] = (k < 92) ? (_Float16)src[(size_t)r * 92 + k] : (_Float16)0.f;
}

// ===========================================================================
// K0c: pad/convert weights to f16; zero the stats/pool region
// ===========================================================================
__global__ void prep_kernel(const float* __restrict__ convW, const float* __restrict__ embW,
                            _Float16* __restrict__ wf16, _Float16* __restrict__ ewf16,
                            float* __restrict__ zr) {
    int i = blockIdx.x * 256 + threadIdx.x;
    if (i < 3 * 128 * 192) {
        int l = i / (128 * 192);
        int rk = i - l * 128 * 192;
        int r = rk / 192, k = rk - r * 192;
        wf16[i] = (k < 169) ? (_Float16)convW[(l * 128 + r) * 169 + k] : (_Float16)0.f;
    }
    if (i < 64 * 96) {
        int r = i / 96, k = i - r * 96;
        ewf16[i] = (k < 92) ? (_Float16)embW[r * 92 + k] : (_Float16)0.f;
    }
    if (i < (int)ZERO_F32) zr[i] = 0.f;
}

// ===========================================================================
// K1: embedding GEMM  x = atom_fea(100000x92) @ emb_W^T(92x64) + emb_b
//     128 rows/block, 4 waves x (2 stripes x 4 col-tiles), K padded to 96.
//     Staging: one row per thread, 12 unrolled b128 copies.
// ===========================================================================
__global__ __launch_bounds__(128) void embed_kernel(
    const _Float16* __restrict__ af16, const _Float16* __restrict__ Wg,
    const float* __restrict__ emb_b, float* __restrict__ xf, _Float16* __restrict__ xh) {
    __shared__ _Float16 Alds[128 * APITCH1];
    const int tid = threadIdx.x;
    const int wave = tid >> 5, lane = tid & 31;
    const int h = lane >> 4, c = lane & 15;
    const int rowblk = blockIdx.x * 128;

    {
        int rg = rowblk + tid;
        half8* d = (half8*)(Alds + tid * APITCH1);
        if (rg < NATOM) {
            const half8* s = (const half8*)(af16 + (size_t)rg * 96);
#pragma unroll
            for (int ch = 0; ch < 12; ++ch) d[ch] = s[ch];
        } else {
            half8 z;
#pragma unroll
            for (int j = 0; j < 8; ++j) z[j] = (_Float16)0.f;
#pragma unroll
            for (int ch = 0; ch < 12; ++ch) d[ch] = z;
        }
    }
    __syncthreads();

    float8 acc[2][4];
#pragma unroll
    for (int s = 0; s < 2; ++s)
#pragma unroll
        for (int nt = 0; nt < 4; ++nt) acc[s][nt] = 0.f;

    const _Float16* arow0 = Alds + (wave * 32 + c) * APITCH1;
    const _Float16* arow1 = Alds + (wave * 32 + 16 + c) * APITCH1;
#pragma unroll
    for (int kc = 0; kc < 3; ++kc) {
        half16 a0 = load_a_frag(arow0, kc * 32, h);
        half16 a1 = load_a_frag(arow1, kc * 32, h);
#pragma unroll
        for (int nt = 0; nt < 4; ++nt) {
            half16 b = load_b_frag(Wg, 96, nt, kc, lane);
            acc[0][nt] = wmma16(a0, b, acc[0][nt]);
            acc[1][nt] = wmma16(a1, b, acc[1][nt]);
        }
    }
#pragma unroll
    for (int s = 0; s < 2; ++s)
#pragma unroll
        for (int nt = 0; nt < 4; ++nt) {
            int col = nt * 16 + c;
            float bias = emb_b[col];
#pragma unroll
            for (int j = 0; j < 8; ++j) {
                int row = rowblk + wave * 32 + s * 16 + 8 * h + j;
                if (row < NATOM) {
                    float v = acc[s][nt][j] + bias;
                    xf[row * 64 + col] = v;
                    xh[row * 64 + col] = (_Float16)v;
                }
            }
        }
}

// ===========================================================================
// K2: conv GEMM pass A.  gated = total(1.2M x 169) @ conv_W[l]^T (169x128).
//     128 rows/block (9375 blocks), 4 waves x 32 rows, K = 6 chunks of 32.
//     Staging: one row per thread, 24 unrolled b128 copies (self | nbr | edge).
//     B-frags stream from L2-resident f16 weight image (reused only 4x/block,
//     so caches beat an LDS copy).  Outputs f16 in accumulator-blocked layout
//     (one b128 store per tile/lane) + per-column sum/sumsq for BN1
//     (conv bias cancels exactly inside training-mode BatchNorm).
// ===========================================================================
__global__ __launch_bounds__(128) void conv_gemm_kernel(
    const _Float16* __restrict__ xh, const _Float16* __restrict__ nbrp,
    const _Float16* __restrict__ Wg, const int* __restrict__ nbr_idx,
    _Float16* __restrict__ gated, float* __restrict__ stats1, int layer) {
    __shared__ _Float16 Alds[128 * APITCH2];   // 51200 B
    __shared__ float stat[256];
    const int tid = threadIdx.x;
    const int wave = tid >> 5, lane = tid & 31;
    const int h = lane >> 4, c = lane & 15;
    const int rowblk = blockIdx.x * 128;

    stat[tid] = 0.f;
    stat[tid + 128] = 0.f;

    // ---- stage A: each thread builds its own 192-half row with b128 copies
    {
        int rg = rowblk + tid;
        int n = rg / 12;
        half8* d = (half8*)(Alds + tid * APITCH2);
        const half8* sx = (const half8*)(xh + (size_t)n * 64);
#pragma unroll
        for (int ch = 0; ch < 8; ++ch) d[ch] = sx[ch];
        int nb = nbr_idx[rg];
        const half8* sn = (const half8*)(xh + (size_t)nb * 64);
#pragma unroll
        for (int ch = 0; ch < 8; ++ch) d[8 + ch] = sn[ch];
        const half8* sf = (const half8*)(nbrp + (size_t)rg * NBRP48);
#pragma unroll
        for (int ch = 0; ch < 6; ++ch) d[16 + ch] = sf[ch];
        half8 z;
#pragma unroll
        for (int j = 0; j < 8; ++j) z[j] = (_Float16)0.f;
        d[22] = z;
        d[23] = z;
    }
    __syncthreads();

    const _Float16* Wl = Wg + layer * (128 * 192);
    float8 acc[2][8];
#pragma unroll
    for (int s = 0; s < 2; ++s)
#pragma unroll
        for (int nt = 0; nt < 8; ++nt) acc[s][nt] = 0.f;

    const _Float16* arow0 = Alds + (wave * 32 + c) * APITCH2;
    const _Float16* arow1 = Alds + (wave * 32 + 16 + c) * APITCH2;
#pragma unroll
    for (int kc = 0; kc < 6; ++kc) {
        half16 a0 = load_a_frag(arow0, kc * 32, h);
        half16 a1 = load_a_frag(arow1, kc * 32, h);
#pragma unroll
        for (int nt = 0; nt < 8; ++nt) {
            half16 b = load_b_frag(Wl, KPAD, nt, kc, lane);
            acc[0][nt] = wmma16(a0, b, acc[0][nt]);
            acc[1][nt] = wmma16(a1, b, acc[1][nt]);
        }
    }

    // ---- stats + blocked f16 store ----
    const int stripe0 = blockIdx.x * 8 + wave * 2;
#pragma unroll
    for (int s = 0; s < 2; ++s)
#pragma unroll
        for (int nt = 0; nt < 8; ++nt) {
            float8 v = acc[s][nt];
            float sm = 0.f, sq = 0.f;
            half8 hv;
#pragma unroll
            for (int j = 0; j < 8; ++j) {
                float x = v[j];
                sm += x;
                sq += x * x;
                hv[j] = (_Float16)x;
            }
            int col = nt * 16 + c;
            atomicAdd(&stat[col], sm);
            atomicAdd(&stat[128 + col], sq);
            *(half8*)(gated + (size_t)(((stripe0 + s) * 8 + nt) * 32 + lane) * 8) = hv;
        }
    __syncthreads();
    atomicAdd(&stats1[layer * 256 + tid], stat[tid]);
    atomicAdd(&stats1[layer * 256 + 128 + tid], stat[tid + 128]);
}

// ===========================================================================
// K4: pass B. Reads blocked gated, applies BN1 affine, sigmoid*softplus,
//     sums the 12 neighbors per atom in LDS (block = 24 stripes = exactly 32
//     atoms via LCM(12,16)=48 alignment -> no global atomics for the sums).
//     Also accumulates per-column BN2 partial stats.
// ===========================================================================
__global__ __launch_bounds__(256) void gate_sum_kernel(
    const _Float16* __restrict__ gated, const float* __restrict__ stats1,
    const float* __restrict__ bn1_g, const float* __restrict__ bn1_b,
    float* __restrict__ nbr_sumed, float* __restrict__ stats2, int layer) {
    __shared__ float accL[32 * 64];
    __shared__ float sc1[128], sh1[128];
    const int tid = threadIdx.x;
    for (int i = tid; i < 2048; i += 256) accL[i] = 0.f;
    if (tid < 128) {
        const float invNM = 1.f / (float)NM;
        float s = stats1[layer * 256 + tid];
        float q = stats1[layer * 256 + 128 + tid];
        float m = s * invNM;
        float var = fmaxf(q * invNM - m * m, 0.f);
        float sc = bn1_g[layer * 128 + tid] * rsqrtf(var + EPS_BN);
        sc1[tid] = sc;
        sh1[tid] = bn1_b[layer * 128 + tid] - m * sc;
    }
    __syncthreads();

    const int atombase = blockIdx.x * 32;
#pragma unroll 1
    for (int it = 0; it < 12; ++it) {
        int task = it * 256 + tid;                 // (stripe_l 0..23, nt 0..3, lane 0..31)
        int gsl = task >> 7;
        int rem = task & 127;
        int nt = rem >> 5, lane = rem & 31;
        int gs = blockIdx.x * 24 + gsl;
        int f = nt * 16 + (lane & 15);
        int baserow = gs * 16 + ((lane >> 4) << 3);
        half8 filt = *(const half8*)(gated + (size_t)((gs * 8 + nt) * 32 + lane) * 8);
        half8 core = *(const half8*)(gated + (size_t)((gs * 8 + nt + 4) * 32 + lane) * 8);
        float scf = sc1[f], shf = sh1[f], scc = sc1[64 + f], shc = sh1[64 + f];
        float partial = 0.f;
        int cura = baserow / 12;
#pragma unroll
        for (int j = 0; j < 8; ++j) {
            int a = (baserow + j) / 12;
            float fn = (float)filt[j] * scf + shf;
            float cn = (float)core[j] * scc + shc;
            float val = sigmoidf(fn) * softplusf(cn);
            if (a != cura) {
                atomicAdd(&accL[(cura - atombase) * 64 + f], partial);
                partial = 0.f;
                cura = a;
            }
            partial += val;
        }
        atomicAdd(&accL[(cura - atombase) * 64 + f], partial);
    }
    __syncthreads();

    if (tid < 64) {  // BN2 partial stats over this block's 32 atoms
        float s = 0.f, q = 0.f;
        for (int a = 0; a < 32; ++a) {
            float v = accL[a * 64 + tid];
            s += v;
            q += v * v;
        }
        atomicAdd(&stats2[layer * 128 + tid], s);
        atomicAdd(&stats2[layer * 128 + 64 + tid], q);
    }
    float4* dst = (float4*)(nbr_sumed + (size_t)atombase * 64);
    const float4* srcv = (const float4*)accL;
    dst[tid] = srcv[tid];
    dst[tid + 256] = srcv[tid + 256];
}

// ===========================================================================
// K6: x = softplus(x + BN2(nbr_sumed));  refresh f32 master + f16 gather copy
// ===========================================================================
__global__ void bn2_res_kernel(const float* __restrict__ ns, const float* __restrict__ stats2,
                               const float* __restrict__ bn2_g, const float* __restrict__ bn2_b,
                               float* __restrict__ xf, _Float16* __restrict__ xh, int layer) {
    int i = blockIdx.x * 256 + threadIdx.x;
    if (i >= NATOM * 64) return;
    int f = i & 63;
    const float invN = 1.f / (float)NATOM;
    float s = stats2[layer * 128 + f];
    float q = stats2[layer * 128 + 64 + f];
    float m = s * invN;
    float var = fmaxf(q * invN - m * m, 0.f);
    float sc = bn2_g[layer * 64 + f] * rsqrtf(var + EPS_BN);
    float sh = bn2_b[layer * 64 + f] - m * sc;
    float v = softplusf(xf[i] + ns[i] * sc + sh);
    xf[i] = v;
    xh[i] = (_Float16)v;
}

// ===========================================================================
// K7: per-crystal pooling (sums + counts) via f32 atomics (L2-resident buffer)
// ===========================================================================
__global__ void pool_kernel(const float* __restrict__ xf, const int* __restrict__ cids,
                            float* __restrict__ csum, float* __restrict__ ccnt) {
    int i = blockIdx.x * 256 + threadIdx.x;
    if (i >= NATOM * 64) return;
    int n = i >> 6, f = i & 63;
    int cid = cids[n];
    atomicAdd(&csum[cid * 64 + f], xf[i]);
    if (f == 0) atomicAdd(&ccnt[cid], 1.f);
}

// ===========================================================================
// K8: head. one 128-thread block per crystal: sp -> fc1(128x80) -> sp -> out
// ===========================================================================
__global__ __launch_bounds__(128) void head_kernel(
    const float* __restrict__ csum, const float* __restrict__ ccnt,
    const float* __restrict__ gfea, const float* __restrict__ fc1W,
    const float* __restrict__ fc1b, const float* __restrict__ outW,
    const float* __restrict__ outb, float* __restrict__ out) {
    __shared__ float v[80];
    __shared__ float hbuf[128];
    __shared__ float red[128];
    const int cgid = blockIdx.x, t = threadIdx.x;
    if (t < 64)
        v[t] = softplusf(csum[cgid * 64 + t] / fmaxf(ccnt[cgid], 1.f));
    else if (t < 80)
        v[t] = softplusf(gfea[cgid * 16 + (t - 64)]);
    __syncthreads();
    float a = fc1b[t];
    for (int j = 0; j < 80; ++j) a += fc1W[t * 80 + j] * v[j];
    hbuf[t] = softplusf(a);
    __syncthreads();
    red[t] = outW[t] * hbuf[t];
    __syncthreads();
    for (int sdist = 64; sdist > 0; sdist >>= 1) {
        if (t < sdist) red[t] += red[t + sdist];
        __syncthreads();
    }
    if (t == 0) out[cgid] = red[0] + outb[0];
}

// ===========================================================================
extern "C" void kernel_launch(void* const* d_in, const int* in_sizes, int n_in,
                              void* d_out, int out_size, void* d_ws, size_t ws_size,
                              hipStream_t stream) {
    const float* atom_fea = (const float*)d_in[0];
    const float* nbr_fea  = (const float*)d_in[1];
    const float* glob_fea = (const float*)d_in[2];
    const float* emb_W    = (const float*)d_in[3];
    const float* emb_b    = (const float*)d_in[4];
    const float* conv_W   = (const float*)d_in[5];
    // d_in[6] conv_b: cancels exactly inside training-mode BatchNorm1d -> unused
    const float* bn1_g    = (const float*)d_in[7];
    const float* bn1_b    = (const float*)d_in[8];
    const float* bn2_g    = (const float*)d_in[9];
    const float* bn2_b    = (const float*)d_in[10];
    const float* fc1_W    = (const float*)d_in[11];
    const float* fc1_b    = (const float*)d_in[12];
    const float* out_W    = (const float*)d_in[13];
    const float* out_b    = (const float*)d_in[14];
    const int*   nbr_idx  = (const int*)d_in[15];
    const int*   cids     = (const int*)d_in[16];
    float* out = (float*)d_out;

    char* ws = (char*)d_ws;
    float*     stats1 = (float*)(ws + OFF_STATS1);
    float*     stats2 = (float*)(ws + OFF_STATS2);
    float*     csum   = (float*)(ws + OFF_CSUM);
    float*     ccnt   = (float*)(ws + OFF_CCNT);
    float*     xf     = (float*)(ws + OFF_XF32);
    _Float16*  xh     = (_Float16*)(ws + OFF_XF16);
    float*     nsum   = (float*)(ws + OFF_NSUM);
    _Float16*  wf16   = (_Float16*)(ws + OFF_WF16);
    _Float16*  ewf16  = (_Float16*)(ws + OFF_EMBW);
    _Float16*  af16   = (_Float16*)(ws + OFF_AF16);
    _Float16*  nbrp   = (_Float16*)(ws + OFF_NBRP);
    _Float16*  gated  = (_Float16*)(ws + OFF_GATED);

    cvt_nbr_pad_kernel<<<225000, 256, 0, stream>>>(nbr_fea, nbrp);   // 57.6M elems
    cvt_atom_pad_kernel<<<37500, 256, 0, stream>>>(atom_fea, af16);  // 9.6M elems
    prep_kernel<<<513, 256, 0, stream>>>(conv_W, emb_W, wf16, ewf16, stats1);
    embed_kernel<<<782, 128, 0, stream>>>(af16, ewf16, emb_b, xf, xh);

    for (int layer = 0; layer < 3; ++layer) {
        conv_gemm_kernel<<<9375, 128, 0, stream>>>(xh, nbrp, wf16, nbr_idx, gated,
                                                   stats1, layer);
        gate_sum_kernel<<<3125, 256, 0, stream>>>(gated, stats1, bn1_g, bn1_b, nsum,
                                                  stats2, layer);
        bn2_res_kernel<<<25000, 256, 0, stream>>>(nsum, stats2, bn2_g, bn2_b, xf, xh,
                                                  layer);
    }

    pool_kernel<<<25000, 256, 0, stream>>>(xf, cids, csum, ccnt);
    head_kernel<<<2000, 128, 0, stream>>>(csum, ccnt, glob_fea, fc1_W, fc1_b, out_W,
                                          out_b, out);
}
